// AttGraphCore_45329084842169
// MI455X (gfx1250) — matrix operand order repaired and checked by reading, not compiled, not run on hardware
//
#include <hip/hip_runtime.h>
#include <math.h>

typedef __bf16 bf16;
typedef __attribute__((ext_vector_type(16))) __bf16 v16bf;
typedef __attribute__((ext_vector_type(8)))  __bf16 v8bfv;
typedef __attribute__((ext_vector_type(8)))  float  v8f;

#define DEV __device__ __forceinline__

DEV bf16  f2b(float x) { return (bf16)x; }
DEV float b2f(bf16 x)  { return (float)x; }

// ---------------------------------------------------------------------------
// WMMA fragment loaders (contiguous k order, same for A and B, wave32)
// A 16x32 (MxK): lane m = lane&15; lanes<16 hold K 0..15, lanes>=16 hold K 16..31
DEV v16bf load_frag_a(const bf16* __restrict__ base, long lda, int k0, int lane) {
  const bf16* p = base + (long)(lane & 15) * lda + k0 + ((lane >> 4) << 4);
  v8bfv lo = *(const v8bfv*)p;          // k = base .. base+7
  v8bfv hi = *(const v8bfv*)(p + 8);    // k = base+8 .. base+15
  v16bf r;
#pragma unroll
  for (int e = 0; e < 8; ++e) { r[e] = lo[e]; r[e + 8] = hi[e]; }
  return r;
}
// B 32x16 (KxN) from "W-format" W (N rows x K cols, row-major):
// lane n = lane&15 ; lanes<16 hold K 0..15, lanes>=16 hold K 16..31 (contiguous in W's row)
DEV v16bf load_frag_b(const bf16* __restrict__ base, long ldw, int k0, int lane) {
  const bf16* p = base + (long)(lane & 15) * ldw + k0 + ((lane >> 4) << 4);
  v8bfv lo = *(const v8bfv*)p;
  v8bfv hi = *(const v8bfv*)(p + 8);
  v16bf r;
#pragma unroll
  for (int e = 0; e < 8; ++e) { r[e] = lo[e]; r[e + 8] = hi[e]; }
  return r;
}

// ---------------------------------------------------------------------------
// Universal W-format WMMA GEMM:  C[M,N] = A[M,K] (bf16) * W[N,K]^T (bf16)
// block = 256 threads = 8 waves (2 Mx4 N) -> block tile 32 (M) x 256 (N);
// each wave: 16x64 (4 accumulators, A fragment amortized over 4 WMMAs).
// Optional: +bias1[n]+bias2[n], relu*mask[row], f32 out, bf16 out, bf16 transposed out.
__global__ __launch_bounds__(256)
void gemm_wmma_bf16(const bf16* __restrict__ A, long sA, int lda,
                    const bf16* __restrict__ W, long sW, int ldw,
                    float* Cf, long sCf, int ldc,
                    bf16* Cb, long sCb, int ldcb,
                    bf16* CbT, long sT, int ldt, int rowsPerBatch,
                    const float* __restrict__ bias1, const float* __restrict__ bias2,
                    const float* __restrict__ mask, int K, int reluMask)
{
  const int z = blockIdx.z;
  A += (long)z * sA;
  W += (long)z * sW;
  const int tid = threadIdx.x, lane = tid & 31, wave = tid >> 5;
  const int wm = wave & 1, wn = wave >> 1;
  const long row0 = (long)blockIdx.y * 32 + wm * 16;
  const int  col0 = blockIdx.x * 256 + wn * 64;
  const bf16* Arow = A + row0 * (long)lda;
  const bf16* W0 = W + (long)(col0 +  0) * ldw;
  const bf16* W1 = W + (long)(col0 + 16) * ldw;
  const bf16* W2 = W + (long)(col0 + 32) * ldw;
  const bf16* W3 = W + (long)(col0 + 48) * ldw;
  v8f acc0 = {0.f,0.f,0.f,0.f,0.f,0.f,0.f,0.f};
  v8f acc1 = {0.f,0.f,0.f,0.f,0.f,0.f,0.f,0.f};
  v8f acc2 = {0.f,0.f,0.f,0.f,0.f,0.f,0.f,0.f};
  v8f acc3 = {0.f,0.f,0.f,0.f,0.f,0.f,0.f,0.f};
  for (int k0 = 0; k0 < K; k0 += 32) {
    v16bf a  = load_frag_a(Arow, lda, k0, lane);
    v16bf b0 = load_frag_b(W0, ldw, k0, lane);
    v16bf b1 = load_frag_b(W1, ldw, k0, lane);
    v16bf b2 = load_frag_b(W2, ldw, k0, lane);
    v16bf b3 = load_frag_b(W3, ldw, k0, lane);
    acc0 = __builtin_amdgcn_wmma_f32_16x16x32_bf16(false, a, false, b0, (short)0, acc0, false, false);
    acc1 = __builtin_amdgcn_wmma_f32_16x16x32_bf16(false, a, false, b1, (short)0, acc1, false, false);
    acc2 = __builtin_amdgcn_wmma_f32_16x16x32_bf16(false, a, false, b2, (short)0, acc2, false, false);
    acc3 = __builtin_amdgcn_wmma_f32_16x16x32_bf16(false, a, false, b3, (short)0, acc3, false, false);
  }
  // C/D layout: element r -> row = row0 + r + 8*(lane>=16), col = tile_col + (lane&15)
  const int nl = lane & 15;
  const int rb = (lane >> 4) << 3;
#pragma unroll
  for (int r = 0; r < 8; ++r) {
    long m = row0 + rb + r;
#pragma unroll
    for (int tgt = 0; tgt < 4; ++tgt) {
      int n = col0 + tgt * 16 + nl;
      float v = (tgt == 0) ? acc0[r] : (tgt == 1) ? acc1[r] : (tgt == 2) ? acc2[r] : acc3[r];
      if (bias1) v += bias1[n];
      if (bias2) v += bias2[n];
      if (reluMask) v = fmaxf(v, 0.f) * mask[m];
      if (Cf) Cf[(long)z * sCf + m * (long)ldc + n] = v;
      if (Cb) Cb[(long)z * sCb + m * (long)ldcb + n] = f2b(v);
      if (CbT) {
        long bt = m / rowsPerBatch;
        long rl = m - bt * rowsPerBatch;
        CbT[bt * sT + (long)n * ldt + rl] = f2b(v);
      }
    }
  }
}

// ---------------------------------------------------------------------------
// Elementwise / prep kernels
__global__ void k_f32_to_bf16(const float* __restrict__ in, bf16* __restrict__ out, long n) {
  long i = (long)blockIdx.x * 256 + threadIdx.x;
  if (i < n) out[i] = f2b(in[i]);
}
// out (rowsOut x colsOut) = in^T, in is (colsOut x rowsOut), f32 -> bf16
__global__ void k_transpose_bf16(const float* __restrict__ in, bf16* __restrict__ out,
                                 int rowsOut, int colsOut) {
  long i = (long)blockIdx.x * 256 + threadIdx.x;
  long tot = (long)rowsOut * colsOut;
  if (i < tot) {
    int n = (int)(i / colsOut);
    int k = (int)(i % colsOut);
    out[i] = f2b(in[(long)k * rowsOut + n]);
  }
}
// Pack W = [W_ih | W_hh] along K, converting to bf16 (LSTM fused weights)
__global__ void k_pack_w2(const float* __restrict__ w1, int K1,
                          const float* __restrict__ w2, int K2,
                          bf16* __restrict__ out, int Nw) {
  int Kt = K1 + K2;
  long i = (long)blockIdx.x * 256 + threadIdx.x;
  if (i < (long)Nw * Kt) {
    int n = (int)(i / Kt);
    int k = (int)(i % Kt);
    float v = (k < K1) ? w1[(long)n * K1 + k] : w2[(long)n * K2 + (k - K1)];
    out[i] = f2b(v);
  }
}
__global__ void k_pack_bias2(const float* __restrict__ b1, const float* __restrict__ b2,
                             float* __restrict__ out) {
  int i = blockIdx.x * 256 + threadIdx.x;   // 2048
  out[i] = (i < 1024) ? b1[i] : b2[i - 1024];
}
// A_att = [prev_h | fc | xt | state_h0]  (64 x 4096), bf16
__global__ void k_pack_a_att(const float* __restrict__ xt, const float* __restrict__ fc,
                             const float* __restrict__ sh, bf16* __restrict__ out) {
  int i = blockIdx.x * 256 + threadIdx.x;   // 64*4096
  int b = i >> 12, k = i & 4095;
  float v;
  if (k < 1024)       v = sh[64 * 1024 + b * 1024 + k];        // prev_h = state_h[1]
  else if (k < 2048)  v = fc[b * 1024 + (k - 1024)];
  else if (k < 3072)  v = xt[b * 1024 + (k - 2048)];
  else                v = sh[b * 1024 + (k - 3072)];           // state_h[0]
  out[i] = f2b(v);
}
__global__ void k_lstm(const float* __restrict__ gates, const float* __restrict__ c_in,
                       float* __restrict__ h_out, float* __restrict__ c_out) {
  int i = blockIdx.x * 256 + threadIdx.x;   // 64*1024
  int b = i >> 10, r = i & 1023;
  const float* g = gates + (long)b * 4096;
  float ig = 1.f / (1.f + expf(-g[r]));
  float fg = 1.f / (1.f + expf(-g[1024 + r]));
  float gg = tanhf(g[2048 + r]);
  float og = 1.f / (1.f + expf(-g[3072 + r]));
  float c = fg * c_in[i] + ig * gg;
  h_out[i] = og * tanhf(c);
  c_out[i] = c;
}
// inp_att bf16 (32768 x 2048) = [graph_embed | h_att] * mask
__global__ void k_build_inp(const float* __restrict__ ge, const float* __restrict__ hv,
                            const float* __restrict__ mask, bf16* __restrict__ out) {
  long i = (long)blockIdx.x * 256 + threadIdx.x;  // 32768*2048
  long rn = i >> 11; int k = (int)(i & 2047); int b = (int)(rn >> 9);
  float v = (k < 1024) ? ge[rn * 1024 + k] : hv[b * 1024 + (k - 1024)];
  out[i] = f2b(v * mask[rn]);
}
// x0^T bf16 per batch (2048 x 512): x0 = [graph_embed | prev_h] * mask
__global__ void k_build_xT(const float* __restrict__ ge, const float* __restrict__ sh,
                           const float* __restrict__ mask, bf16* __restrict__ xT) {
  long i = (long)blockIdx.x * 256 + threadIdx.x;  // 32768*2048
  long rn = i >> 11; int k = (int)(i & 2047);
  int b = (int)(rn >> 9); int n = (int)(rn & 511);
  float v = (k < 1024) ? ge[rn * 1024 + k] : sh[64 * 1024 + b * 1024 + (k - 1024)];
  xT[(long)b * (2048 * 512) + (long)k * 512 + n] = f2b(v * mask[rn]);
}
__global__ __launch_bounds__(256) void k_batch_min(const float* __restrict__ adj,
                                                   float* __restrict__ minb) {
  __shared__ float red[256];
  int b = blockIdx.x;
  const float* p = adj + (long)b * 512 * 512;
  float m = 3.402823466e38f;
  for (int j = threadIdx.x; j < 512 * 512; j += 256) m = fminf(m, p[j]);
  red[threadIdx.x] = m; __syncthreads();
  for (int s = 128; s > 0; s >>= 1) {
    if (threadIdx.x < s) red[threadIdx.x] = fminf(red[threadIdx.x], red[threadIdx.x + s]);
    __syncthreads();
  }
  if (threadIdx.x == 0) minb[b] = red[0];
}
// Per-row: v=(adj-min_b)*mask_row; stable top-256 of 512 via bitonic sort on (v desc, idx asc);
// zero the rest in place; d_row = rowsum>0 ? rsqrt(rowsum) : 0
__global__ __launch_bounds__(256) void k_topk(float* __restrict__ adj,
                                              const float* __restrict__ minb,
                                              const float* __restrict__ mask,
                                              float* __restrict__ dout) {
  __shared__ float sv[512];
  __shared__ short si[512];
  __shared__ float orig[512];
  __shared__ float red[256];
  int row = blockIdx.x; int b = row >> 9;
  float* ar = adj + (long)row * 512;
  float mb = minb[b], mr = mask[row];
  int t = threadIdx.x;
  for (int j = t; j < 512; j += 256) {
    float v = (ar[j] - mb) * mr;
    orig[j] = v; sv[j] = v; si[j] = (short)j;
  }
  __syncthreads();
  for (int size = 2; size <= 512; size <<= 1) {
    for (int stride = size >> 1; stride > 0; stride >>= 1) {
      int lo = 2 * stride * (t / stride) + (t % stride);
      int hi = lo + stride;
      bool desc = ((lo & size) == 0);
      float vl = sv[lo], vh = sv[hi];
      short il = si[lo], ih = si[hi];
      bool hiFirst = (vh > vl) || (vh == vl && ih < il);
      if (desc == hiFirst) { sv[lo] = vh; sv[hi] = vl; si[lo] = ih; si[hi] = il; }
      __syncthreads();
    }
  }
  float kv = sv[255]; short ki = si[255];   // K = 256
  float local = 0.f;
  for (int j = t; j < 512; j += 256) {
    float v = orig[j];
    bool keep = (v > kv) || (v == kv && j <= ki);
    float o = keep ? v : 0.f;
    ar[j] = o; local += o;
  }
  red[t] = local; __syncthreads();
  for (int s = 128; s > 0; s >>= 1) {
    if (t < s) red[t] += red[t + s];
    __syncthreads();
  }
  if (t == 0) { float rs = red[0]; dout[row] = (rs > 0.f) ? rsqrtf(rs) : 0.f; }
}
__global__ void k_adj_bf16(const float* __restrict__ adj, const float* __restrict__ d,
                           bf16* __restrict__ out) {
  long i = (long)blockIdx.x * 256 + threadIdx.x;  // 64*512*512
  int b = (int)(i >> 18), rr = (int)((i >> 9) & 511), cc = (int)(i & 511);
  out[i] = f2b(adj[i] * d[b * 512 + rr] * d[b * 512 + cc]);
}
// att[b,r] = max_n x3[b,n,r]; write into A_lang cols [0,1024)
__global__ void k_att_max(const bf16* __restrict__ x3, bf16* __restrict__ Alang) {
  int i = blockIdx.x * 256 + threadIdx.x;  // 64*1024
  int b = i >> 10, r = i & 1023;
  const bf16* p = x3 + (long)b * 512 * 1024 + r;
  float m = b2f(p[0]);
  for (int n = 1; n < 512; ++n) m = fmaxf(m, b2f(p[(long)n * 1024]));
  Alang[(long)b * 3072 + r] = f2b(m);
}
__global__ void k_pack_lang(const float* __restrict__ h_att, const float* __restrict__ sh,
                            bf16* __restrict__ Alang) {
  int i = blockIdx.x * 256 + threadIdx.x;  // 64*2048
  int b = i >> 11, k = i & 2047;
  float v = (k < 1024) ? h_att[b * 1024 + k] : sh[64 * 1024 + b * 1024 + (k - 1024)];
  Alang[(long)b * 3072 + 1024 + k] = f2b(v);
}
__global__ void k_out(const float* __restrict__ h_att, const float* __restrict__ c_att,
                      const float* __restrict__ h_l, const float* __restrict__ c_l,
                      float* __restrict__ out) {
  int i = blockIdx.x * 256 + threadIdx.x;  // 64*1024
  out[i] = h_l[i];                 // output
  out[65536 + i] = h_att[i];       // stacked h[0]
  out[131072 + i] = h_l[i];        // stacked h[1]
  out[196608 + i] = c_att[i];      // stacked c[0]
  out[262144 + i] = c_l[i];        // stacked c[1]
}

// ---------------------------------------------------------------------------
extern "C" void kernel_launch(void* const* d_in, const int* in_sizes, int n_in,
                              void* d_out, int out_size, void* d_ws, size_t ws_size,
                              hipStream_t stream) {
  const float* xt    = (const float*)d_in[0];
  const float* fc    = (const float*)d_in[1];
  const float* ge    = (const float*)d_in[2];
  const float* sh    = (const float*)d_in[4];   // state_h (2,64,1024)
  const float* sc    = (const float*)d_in[5];   // state_c (2,64,1024)
  const float* msk   = (const float*)d_in[6];   // att_mask (64,512)
  const float* aWih  = (const float*)d_in[7];
  const float* aWhh  = (const float*)d_in[8];
  const float* aBih  = (const float*)d_in[9];
  const float* aBhh  = (const float*)d_in[10];
  const float* lWih  = (const float*)d_in[11];
  const float* lWhh  = (const float*)d_in[12];
  const float* lBih  = (const float*)d_in[13];
  const float* lBhh  = (const float*)d_in[14];
  const float* e1W   = (const float*)d_in[15];
  const float* e1b   = (const float*)d_in[16];
  const float* e2W   = (const float*)d_in[17];
  const float* e2b   = (const float*)d_in[18];
  const float* g0W   = (const float*)d_in[19];
  const float* g0b   = (const float*)d_in[20];
  const float* g1W   = (const float*)d_in[21];
  const float* g1b   = (const float*)d_in[22];
  const float* g2W   = (const float*)d_in[23];
  const float* g2b   = (const float*)d_in[24];
  float* out = (float*)d_out;

  char* ws = (char*)d_ws;
  size_t o = 0;
  auto alloc = [&](size_t bytes) { size_t r = o; o = (o + bytes + 255) & ~(size_t)255; return r; };
  const long MN = 32768;  // B*N
  bf16* W_ATT  = (bf16*)(ws + alloc((size_t)4096 * 4096 * 2));
  bf16* W_LANG = (bf16*)(ws + alloc((size_t)4096 * 3072 * 2));
  bf16* W_E12  = (bf16*)(ws + alloc((size_t)2048 * 2048 * 2));   // [emb1_W ; emb2_W]
  bf16* W_G0T  = (bf16*)(ws + alloc((size_t)1024 * 2048 * 2));
  bf16* W_G1T  = (bf16*)(ws + alloc((size_t)1024 * 1024 * 2));
  bf16* W_G2T  = (bf16*)(ws + alloc((size_t)1024 * 1024 * 2));
  bf16* A_ATT  = (bf16*)(ws + alloc((size_t)64 * 4096 * 2));
  bf16* A_LANG = (bf16*)(ws + alloc((size_t)64 * 3072 * 2));
  bf16* BIGBF  = (bf16*)(ws + alloc((size_t)MN * 2048 * 2));   // inp_att, then x3
  bf16* EC     = (bf16*)(ws + alloc((size_t)MN * 2048 * 2));   // [e1|e2] per row; later t
  bf16* T      = EC;                                           // t buffer aliases EC
  bf16* XT     = (bf16*)(ws + alloc((size_t)MN * 2048 * 2));   // x^T per batch
  bf16* ADJB   = (bf16*)(ws + alloc((size_t)64 * 512 * 512 * 2));
  float* ADJF  = (float*)(ws + alloc((size_t)64 * 512 * 512 * 4));
  float* GATES = (float*)(ws + alloc((size_t)64 * 4096 * 4));
  float* HATT  = (float*)(ws + alloc((size_t)64 * 1024 * 4));
  float* CATT  = (float*)(ws + alloc((size_t)64 * 1024 * 4));
  float* HL    = (float*)(ws + alloc((size_t)64 * 1024 * 4));
  float* CL    = (float*)(ws + alloc((size_t)64 * 1024 * 4));
  float* DV    = (float*)(ws + alloc((size_t)MN * 4));
  float* MINB  = (float*)(ws + alloc((size_t)64 * 4));
  float* EB12  = (float*)(ws + alloc((size_t)2048 * 4));

  auto blks = [](long n) { return (unsigned)((n + 255) / 256); };

  // --- weight prep (bf16 pack / transpose) ---
  k_pack_w2<<<blks((long)4096 * 4096), 256, 0, stream>>>(aWih, 3072, aWhh, 1024, W_ATT, 4096);
  k_pack_w2<<<blks((long)4096 * 3072), 256, 0, stream>>>(lWih, 2048, lWhh, 1024, W_LANG, 4096);
  k_f32_to_bf16<<<blks((long)1024 * 2048), 256, 0, stream>>>(e1W, W_E12, (long)1024 * 2048);
  k_f32_to_bf16<<<blks((long)1024 * 2048), 256, 0, stream>>>(e2W, W_E12 + (long)1024 * 2048,
                                                             (long)1024 * 2048);
  k_pack_bias2<<<8, 256, 0, stream>>>(e1b, e2b, EB12);
  k_transpose_bf16<<<blks((long)1024 * 2048), 256, 0, stream>>>(g0W, W_G0T, 1024, 2048);
  k_transpose_bf16<<<blks((long)1024 * 1024), 256, 0, stream>>>(g1W, W_G1T, 1024, 1024);
  k_transpose_bf16<<<blks((long)1024 * 1024), 256, 0, stream>>>(g2W, W_G2T, 1024, 1024);

  // --- attention LSTM: gates = [prev_h|fc|xt|h0] @ [W_ih|W_hh]^T + b_ih + b_hh ---
  k_pack_a_att<<<blks((long)64 * 4096), 256, 0, stream>>>(xt, fc, sh, A_ATT);
  gemm_wmma_bf16<<<dim3(4096 / 256, 64 / 32, 1), 256, 0, stream>>>(
      A_ATT, 0, 4096, W_ATT, 0, 4096,
      GATES, 0, 4096, nullptr, 0, 0, nullptr, 0, 0, 1,
      aBih, aBhh, nullptr, 4096, 0);
  k_lstm<<<blks((long)64 * 1024), 256, 0, stream>>>(GATES, sc, HATT, CATT);

  // --- fused e1|e2 projection: EC = inp_att @ [emb1_W;emb2_W]^T + [b1|b2] ---
  k_build_inp<<<blks(MN * 2048), 256, 0, stream>>>(ge, HATT, msk, BIGBF);
  gemm_wmma_bf16<<<dim3(2048 / 256, MN / 32, 1), 256, 0, stream>>>(
      BIGBF, 0, 2048, W_E12, 0, 2048,
      nullptr, 0, 0, EC, 0, 2048, nullptr, 0, 0, 1,
      EB12, nullptr, nullptr, 2048, 0);

  // --- Adj = e1 @ e2^T (batched; A = EC cols[0:1024), W-format B = EC cols[1024:2048)) ---
  gemm_wmma_bf16<<<dim3(512 / 256, 512 / 32, 64), 256, 0, stream>>>(
      EC, (long)512 * 2048, 2048, EC + 1024, (long)512 * 2048, 2048,
      ADJF, (long)512 * 512, 512, nullptr, 0, 0, nullptr, 0, 0, 1,
      nullptr, nullptr, nullptr, 1024, 0);

  // --- min, top-K + degree normalize, bf16 convert ---
  k_batch_min<<<64, 256, 0, stream>>>(ADJF, MINB);
  k_topk<<<32768, 256, 0, stream>>>(ADJF, MINB, msk, DV);
  k_adj_bf16<<<blks((long)64 * 512 * 512), 256, 0, stream>>>(ADJF, DV, ADJB);

  // --- GCN input x0^T ---
  k_build_xT<<<blks(MN * 2048), 256, 0, stream>>>(ge, sh, msk, XT);

  // --- GCN layer 0: t = Adj@x0 ; x1 = relu(t@W0 + b0)*m ---
  gemm_wmma_bf16<<<dim3(2048 / 256, 512 / 32, 64), 256, 0, stream>>>(
      ADJB, (long)512 * 512, 512, XT, (long)2048 * 512, 512,
      nullptr, 0, 0, T, (long)512 * 2048, 2048, nullptr, 0, 0, 1,
      nullptr, nullptr, nullptr, 512, 0);
  gemm_wmma_bf16<<<dim3(1024 / 256, MN / 32, 1), 256, 0, stream>>>(
      T, 0, 2048, W_G0T, 0, 2048,
      nullptr, 0, 0, nullptr, 0, 0, XT, (long)1024 * 512, 512, 512,
      g0b, nullptr, msk, 2048, 1);

  // --- GCN layer 1 ---
  gemm_wmma_bf16<<<dim3(1024 / 256, 512 / 32, 64), 256, 0, stream>>>(
      ADJB, (long)512 * 512, 512, XT, (long)1024 * 512, 512,
      nullptr, 0, 0, T, (long)512 * 1024, 1024, nullptr, 0, 0, 1,
      nullptr, nullptr, nullptr, 512, 0);
  gemm_wmma_bf16<<<dim3(1024 / 256, MN / 32, 1), 256, 0, stream>>>(
      T, 0, 1024, W_G1T, 0, 1024,
      nullptr, 0, 0, nullptr, 0, 0, XT, (long)1024 * 512, 512, 512,
      g1b, nullptr, msk, 1024, 1);

  // --- GCN layer 2 (row-major bf16 x3 out for max-pool) ---
  gemm_wmma_bf16<<<dim3(1024 / 256, 512 / 32, 64), 256, 0, stream>>>(
      ADJB, (long)512 * 512, 512, XT, (long)1024 * 512, 512,
      nullptr, 0, 0, T, (long)512 * 1024, 1024, nullptr, 0, 0, 1,
      nullptr, nullptr, nullptr, 512, 0);
  gemm_wmma_bf16<<<dim3(1024 / 256, MN / 32, 1), 256, 0, stream>>>(
      T, 0, 1024, W_G2T, 0, 1024,
      nullptr, 0, 0, BIGBF, 0, 1024, nullptr, 0, 0, 1,
      g2b, nullptr, msk, 1024, 1);

  // --- att = max_n x3 ; language LSTM ---
  k_att_max<<<blks((long)64 * 1024), 256, 0, stream>>>(BIGBF, A_LANG);
  k_pack_lang<<<blks((long)64 * 2048), 256, 0, stream>>>(HATT, sh, A_LANG);
  gemm_wmma_bf16<<<dim3(4096 / 256, 64 / 32, 1), 256, 0, stream>>>(
      A_LANG, 0, 3072, W_LANG, 0, 3072,
      GATES, 0, 4096, nullptr, 0, 0, nullptr, 0, 0, 1,
      lBih, lBhh, nullptr, 3072, 0);
  k_lstm<<<blks((long)64 * 1024), 256, 0, stream>>>(GATES, sc + (long)64 * 1024, HL, CL);

  // --- outputs: h_lang, stack(h_att,h_lang), stack(c_att,c_lang) ---
  k_out<<<blks((long)64 * 1024), 256, 0, stream>>>(HATT, CATT, HL, CL, out);
}
